// CCS_block_46273977647540
// MI455X (gfx1250) — compile-verified, matrix operand order checked by reading, not run on hardware
//
#include <hip/hip_runtime.h>

typedef __attribute__((ext_vector_type(16))) _Float16 v16h;
typedef __attribute__((ext_vector_type(8)))  _Float16 v8h;
typedef __attribute__((ext_vector_type(2)))  _Float16 v2h;
typedef __attribute__((ext_vector_type(8)))  float    v8f;
typedef __attribute__((ext_vector_type(4)))  float    v4f;

#define BN 100           // tokens per batch
#define BD 768           // feature dim
#define NT 7             // number of 16-row tiles (covers 112 >= 100)
#define KB 24            // 768 / 32 K-steps
#define NTHREADS 256     // 8 waves of 32

__global__ __launch_bounds__(NTHREADS, 1)
void ccs_fused_kernel(const float* __restrict__ x,
                      const float* __restrict__ cc,
                      const float* __restrict__ alpha,
                      const float* __restrict__ ln_gamma,
                      const float* __restrict__ ln_beta,
                      const float* __restrict__ th_w,
                      const float* __restrict__ th_b,
                      float* __restrict__ out)
{
    constexpr float EPS_LN = 1e-5f;
    constexpr float EPS    = 1e-8f;

    // Rows 0..99 = normalized tile; row 100 = dedicated zero row so padded
    // lanes read zeros with NO per-fragment masking VALU work.
    __shared__ __attribute__((aligned(16))) _Float16 xnf[BN + 1][BD]; // 155.1 KB
    __shared__ float nrm[NT * 16];     // row norms (padded rows get 1.0)
    __shared__ float dens[NT * 16];    // density accumulators
    __shared__ float wts[NT * 16];     // relu-gated weights
    __shared__ float sS[1];            // sum of weights

    const int b    = blockIdx.x;
    const int tid  = threadIdx.x;
    const int lane = tid & 31;
    const int wave = tid >> 5;

    // zero density accumulators + the zero row (96 threads x ds_store_b128)
    for (int i = tid; i < NT * 16; i += NTHREADS) dens[i] = 0.0f;
    if (tid < BD / 8) {
        v8h z = {};
        *(v8h*)&xnf[BN][tid * 8] = z;
    }

    const float* xb = x + (size_t)b * BN * BD;

    // ---------------- Stage 1: LayerNorm rows -> f16 LDS + row norms --------
    // Each lane owns 4 contiguous floats x 6 chunks: global_load_b128 path.
    for (int r = wave; r < BN; r += 8) {
        const float* row = xb + (size_t)r * BD;
        __builtin_prefetch(row + 8 * BD, 0, 0);   // global_prefetch_b8
        v4f v[6];
        float s = 0.0f, s2 = 0.0f;
        #pragma unroll
        for (int i = 0; i < 6; ++i) {
            v4f t = *(const v4f*)&row[lane * 4 + 128 * i];
            v[i] = t;
            s  += t[0] + t[1] + t[2] + t[3];
            s2 += t[0]*t[0] + t[1]*t[1] + t[2]*t[2] + t[3]*t[3];
        }
        #pragma unroll
        for (int m = 16; m >= 1; m >>= 1) {
            s  += __shfl_xor(s,  m, 32);
            s2 += __shfl_xor(s2, m, 32);
        }
        const float mu   = s * (1.0f / BD);
        const float var  = s2 * (1.0f / BD) - mu * mu;
        const float rstd = rsqrtf(var + EPS_LN);
        float n2 = 0.0f;
        #pragma unroll
        for (int i = 0; i < 6; ++i) {
            const int d0 = lane * 4 + 128 * i;
            const v4f g  = *(const v4f*)&ln_gamma[d0];
            const v4f be = *(const v4f*)&ln_beta[d0];
            _Float16 h[4];
            #pragma unroll
            for (int j = 0; j < 4; ++j) {
                const float xn = (v[i][j] - mu) * rstd * g[j] + be[j];
                n2 += xn * xn;
                h[j] = (_Float16)xn;
            }
            union { _Float16 hh[4]; unsigned long long u; } pk;
            pk.hh[0] = h[0]; pk.hh[1] = h[1]; pk.hh[2] = h[2]; pk.hh[3] = h[3];
            *(unsigned long long*)&xnf[r][d0] = pk.u;   // ds_store_b64
        }
        #pragma unroll
        for (int m = 16; m >= 1; m >>= 1) n2 += __shfl_xor(n2, m, 32);
        if (lane == 0) nrm[r] = sqrtf(n2);
    }
    if (tid < NT * 16 - BN) nrm[BN + tid] = 1.0f;   // pad norms
    __syncthreads();

    // ---------------- Stage 2: Gram matrix via v_wmma_f32_16x16x32_f16 ------
    // Wave w (< 7) owns tile-row mi = w: hoists the A fragment per k-step and
    // streams 7 B fragments against 7 live accumulators (56 VGPRs).
    // 16-bit A/B lane layout (ISA 7.12.2): lane&15 = matrix row,
    // lane>>4 selects K-chunks {0..7,16..23} vs {8..15,24..31}.
    const int khalf = lane >> 4;
    const int l15   = lane & 15;

    if (wave < NT) {
        const int mi = wave;
        const int ra = mi * 16 + l15;
        const _Float16* arow = xnf[(ra < BN) ? ra : BN];  // clamp to zero row

        const _Float16* brow[NT];
        #pragma unroll
        for (int ni = 0; ni < NT; ++ni) {
            const int rb = ni * 16 + l15;
            brow[ni] = xnf[(rb < BN) ? rb : BN];
        }

        v8f acc[NT];
        #pragma unroll
        for (int ni = 0; ni < NT; ++ni) acc[ni] = (v8f){};

        #pragma unroll 2
        for (int kb = 0; kb < KB; ++kb) {
            const int k0 = kb * 32 + khalf * 8;
            union { v16h v; v8h h[2]; } A;
            A.h[0] = *(const v8h*)&arow[k0];        // ds_load_b128
            A.h[1] = *(const v8h*)&arow[k0 + 16];
            #pragma unroll
            for (int ni = 0; ni < NT; ++ni) {
                union { v16h v; v8h h[2]; } Bm;
                Bm.h[0] = *(const v8h*)&brow[ni][k0];
                Bm.h[1] = *(const v8h*)&brow[ni][k0 + 16];
                acc[ni] = __builtin_amdgcn_wmma_f32_16x16x32_f16(
                              false, A.v, false, Bm.v, (short)0, acc[ni],
                              false, false);
            }
        }

        // C layout: lanes 0-15 N=lane (M=j), lanes 16-31 N=lane-16 (M=8+j)
        #pragma unroll
        for (int ni = 0; ni < NT; ++ni) {
            const float nn = nrm[ni * 16 + l15];
            #pragma unroll
            for (int j = 0; j < 8; ++j) {
                const int m = mi * 16 + khalf * 8 + j;
                float c = acc[ni][j] / (nrm[m] * nn + EPS);
                // sum over N: reduce the 16 lanes of this half-wave
                c += __shfl_xor(c, 1, 32);
                c += __shfl_xor(c, 2, 32);
                c += __shfl_xor(c, 4, 32);
                c += __shfl_xor(c, 8, 32);
                if (l15 == 0) atomicAdd(&dens[m], c);   // ds_add_f32
            }
        }
    }
    __syncthreads();

    // ---------------- Stage 3: min-max norm + learned threshold (wave 0) ----
    if (wave == 0) {
        float dmax = -3.4e38f, dmin = 3.4e38f;
        for (int n = lane; n < BN; n += 32) {
            const float d = dens[n];
            dmax = fmaxf(dmax, d);
            dmin = fminf(dmin, d);
        }
        #pragma unroll
        for (int m = 16; m >= 1; m >>= 1) {
            dmax = fmaxf(dmax, __shfl_xor(dmax, m, 32));
            dmin = fminf(dmin, __shfl_xor(dmin, m, 32));
        }
        const float inv = 1.0f / (dmax - dmin + EPS);
        float dot = 0.0f;
        for (int n = lane; n < BN; n += 32) {
            const float dn = (dens[n] - dmin) * inv;
            dens[n] = dn;
            dot += dn * th_w[n];
        }
        #pragma unroll
        for (int m = 16; m >= 1; m >>= 1) dot += __shfl_xor(dot, m, 32);
        const float th = (1.0f / (1.0f + __expf(-(dot + th_b[0])))) * alpha[0];
        float S = 0.0f;
        for (int n = lane; n < BN; n += 32) {
            const float w = fmaxf(dens[n] - th, 0.0f);
            wts[n] = w;
            S += w;
        }
        #pragma unroll
        for (int m = 16; m >= 1; m >>= 1) S += __shfl_xor(S, m, 32);
        if (lane == 0) sS[0] = S;
    }
    __syncthreads();

    // ---------------- Stage 4: weighted mean shift of cluster center --------
    const float S     = sS[0];
    const float scale = 1.0f / ((S + EPS) * (float)BN);
    const float* ccb  = cc + (size_t)b * BD;
    float* outb       = out + (size_t)b * BD;
    for (int p = tid; p < BD / 2; p += NTHREADS) {
        float a0 = 0.0f, a1 = 0.0f;
        for (int n = 0; n < BN; ++n) {
            const v2h t = *(const v2h*)&xnf[n][2 * p];   // ds_load_b32
            const float w = wts[n];
            a0 += w * (float)t[0];
            a1 += w * (float)t[1];
        }
        const float c0 = ccb[2 * p];
        const float c1 = ccb[2 * p + 1];
        float2 o;
        o.x = c0 + (a0 - c0 * S) * scale;
        o.y = c1 + (a1 - c1 * S) * scale;
        *(float2*)&outb[2 * p] = o;
    }
}

extern "C" void kernel_launch(void* const* d_in, const int* in_sizes, int n_in,
                              void* d_out, int out_size, void* d_ws, size_t ws_size,
                              hipStream_t stream) {
    (void)n_in; (void)d_ws; (void)ws_size; (void)out_size;
    const float* x     = (const float*)d_in[0];
    const float* cc    = (const float*)d_in[1];
    const float* alpha = (const float*)d_in[2];
    const float* gam   = (const float*)d_in[3];
    const float* bet   = (const float*)d_in[4];
    const float* th_w  = (const float*)d_in[5];
    const float* th_b  = (const float*)d_in[6];
    float* out = (float*)d_out;

    const int B = in_sizes[0] / (BN * BD);   // 1024
    ccs_fused_kernel<<<B, NTHREADS, 0, stream>>>(x, cc, alpha, gam, bet,
                                                 th_w, th_b, out);
}
